// MoE_2370821947876
// MI455X (gfx1250) — matrix operand order repaired and checked by reading, not compiled, not run on hardware
//
#include <hip/hip_runtime.h>
#include <math.h>

// ---------------- problem constants (from reference) ----------------
#define NTOK  4096      // B*S = 2*2048
#define H_DIM 1280
#define I_DIM 1280
#define E_NUM 63
#define TOPK  7

// ---------------- WMMA fragment types ----------------
typedef __attribute__((ext_vector_type(16))) __bf16 v16bf;
typedef __attribute__((ext_vector_type(8)))  __bf16 v8bf;
typedef __attribute__((ext_vector_type(2)))  __bf16 v2bf;
typedef __attribute__((ext_vector_type(8)))  float  f32x8;
typedef unsigned short u16;
typedef unsigned int   u32;

// Block tile: 256 threads = 8 waves as 2(M) x 4(N); each wave owns 32x32
// -> block = 64M x 128N, K-step 32, 4 WMMAs per wave per stage (32/block).
// LDS double-buffered: one barrier per k-step.
#define BM 64
#define BN 128
#define BK 32
#define ASTR 40   // LDS k-stride (bf16); 40*2=80B keeps 16B alignment per row
#define BSTR 40

__device__ __forceinline__ u32 pack_bf16(float lo, float hi) {
#if __has_builtin(__builtin_amdgcn_cvt_pk_bf16_f32)
    return __builtin_bit_cast(u32, __builtin_amdgcn_cvt_pk_bf16_f32(lo, hi));
#else
    v2bf v = {(__bf16)lo, (__bf16)hi};
    return __builtin_bit_cast(u32, v);
#endif
}

__device__ __forceinline__ float gelu_exact(float v) {
    return 0.5f * v * (1.0f + erff(v * 0.7071067811865476f));
}

// A fragment (16x32 bf16): lane L holds row L&15,
// K = {kb..kb+7, kb+16..kb+23}, kb=(L>>4)*8   (ISA 16-bit A layout)
__device__ __forceinline__ v16bf load_a_frag(const __bf16* base, int lane) {
    const __bf16* p = base + (lane & 15) * ASTR + ((lane >> 4) * 8);
    v8bf lo = *(const v8bf*)(p);
    v8bf hi = *(const v8bf*)(p + 16);
    v16bf r;
#pragma unroll
    for (int i = 0; i < 8; ++i) { r[i] = lo[i]; r[i + 8] = hi[i]; }
    return r;
}

// B fragment (32x16 bf16): lane L holds col L&15, K = {kb..kb+7, kb+16..kb+23},
// kb=(L>>4)*8. LDS tile stored transposed [n][k].
__device__ __forceinline__ v16bf load_b_frag(const __bf16* base, int lane) {
    const __bf16* p = base + (lane & 15) * BSTR + ((lane >> 4) * 8);
    v8bf lo = *(const v8bf*)(p);
    v8bf hi = *(const v8bf*)(p + 16);
    v16bf r;
#pragma unroll
    for (int i = 0; i < 8; ++i) { r[i] = lo[i]; r[i + 8] = hi[i]; }
    return r;
}

// ---------------- router: softmax + top-7 + expert scatter ----------------
__global__ __launch_bounds__(64)
void router_kernel(const float* __restrict__ x,
                   const float* __restrict__ rw,   // [H, E]
                   const float* __restrict__ rb,   // [E]
                   int*   __restrict__ cnt,
                   int*   __restrict__ tok_list,   // [E][NTOK]
                   float* __restrict__ gate_list)  // [E][NTOK]
{
    int t = blockIdx.x;
    int e = threadIdx.x;
    __shared__ float aff[64];

    float acc = 0.f;
    if (e < E_NUM) {
        const float* xp = x + (size_t)t * H_DIM;
        for (int h = 0; h < H_DIM; ++h)
            acc = fmaf(xp[h], rw[h * E_NUM + e], acc);   // rw coalesced over e
        acc += rb[e];
    }
    aff[e] = (e < E_NUM) ? acc : -1e30f;
    __syncthreads();

    if (e == 0) {
        float m = -1e30f;
        for (int i = 0; i < E_NUM; ++i) m = fmaxf(m, aff[i]);
        float s = 0.f;
        for (int i = 0; i < E_NUM; ++i) { float ev = expf(aff[i] - m); aff[i] = ev; s += ev; }
        float inv = 1.f / s;
        for (int i = 0; i < E_NUM; ++i) aff[i] *= inv;
        for (int k = 0; k < TOPK; ++k) {       // earliest index wins ties
            int bi = 0; float bv = aff[0];
            for (int i = 1; i < E_NUM; ++i) if (aff[i] > bv) { bv = aff[i]; bi = i; }
            aff[bi] = -1.f;
            int pos = atomicAdd(&cnt[bi], 1);
            tok_list[bi * NTOK + pos]  = t;
            gate_list[bi * NTOK + pos] = bv;
        }
    }
}

__global__ void offsets_kernel(const int* __restrict__ cnt, int* __restrict__ offsets) {
    if (threadIdx.x == 0) {
        int s = 0;
        for (int i = 0; i < E_NUM; ++i) { offsets[i] = s; s += cnt[i]; }
        offsets[E_NUM] = s;
    }
}

// ---------------- GEMM1: Hbuf = gelu(X[gather] @ W1 + b1), bf16 out ----------------
template <bool ROUTED>
__global__ __launch_bounds__(256)
void gemm1_kernel(const float* __restrict__ X,     // [NTOK, H]
                  const float* __restrict__ W1,    // [(E,)H, I]
                  const float* __restrict__ B1,    // [(E,)I]
                  u16*         __restrict__ Hbuf,  // [rows_total, I] bf16 bits
                  const int*   __restrict__ tok_list,
                  const int*   __restrict__ cnt,
                  const int*   __restrict__ offsets)
{
    const int e    = ROUTED ? (int)blockIdx.z : 0;
    const int rows = ROUTED ? cnt[e] : NTOK;
    const int m0   = blockIdx.x * BM;
    if (m0 >= rows) return;
    const int n0   = blockIdx.y * BN;

    const float* W    = W1 + (size_t)e * H_DIM * I_DIM;
    const float* bias = B1 + (size_t)e * I_DIM;
    const size_t hbase = ROUTED ? (size_t)offsets[e] : 0;

    __shared__ __align__(16) __bf16 Alds[2][BM * ASTR];   // [m][k], double buffer
    __shared__ __align__(16) __bf16 Blds[2][BN * BSTR];   // [n][k] transposed

    const int tid  = threadIdx.x;
    const int lane = tid & 31;
    const int wave = tid >> 5;
    const int wm   = wave >> 2;   // 0..1 -> rows wm*32
    const int wn   = wave & 3;    // 0..3 -> cols wn*32

    // ---- A loader: 2 units/thread, unit = float4 along k ----
    int  a_r[2], a_c4[2];
    const float* a_ptr[2];
    bool a_v[2];
#pragma unroll
    for (int q = 0; q < 2; ++q) {
        int idx = tid + q * 256;
        a_r[q]  = idx >> 3;            // 0..63
        a_c4[q] = (idx & 7) * 4;       // 0..28
        int gr  = m0 + a_r[q];
        a_v[q]  = gr < rows;
        int tok = 0;
        if (a_v[q]) tok = ROUTED ? tok_list[e * NTOK + gr] : gr;
        a_ptr[q] = X + (size_t)tok * H_DIM + a_c4[q];
    }
    // ---- B loader: 2 units/thread, unit = 2 k-rows x 4 n (coalesced over n) ----
    int b_k[2], b_n[2];
#pragma unroll
    for (int q = 0; q < 2; ++q) {
        int idx = tid + q * 256;
        b_n[q] = (idx & 31) * 4;       // 0..124
        b_k[q] = (idx >> 5) * 2;       // 0..30
    }

    float4 aP[2], b0P[2], b1P[2];
    f32x8 acc[2][2];
#pragma unroll
    for (int i = 0; i < 2; ++i)
#pragma unroll
        for (int j = 0; j < 2; ++j)
            acc[i][j] = (f32x8){0.f,0.f,0.f,0.f,0.f,0.f,0.f,0.f};

    auto load_regs = [&](int K0) {
#pragma unroll
        for (int q = 0; q < 2; ++q)
            aP[q] = a_v[q] ? *(const float4*)(a_ptr[q] + K0)
                           : make_float4(0.f, 0.f, 0.f, 0.f);
#pragma unroll
        for (int q = 0; q < 2; ++q) {
            const float* wp = W + (size_t)(K0 + b_k[q]) * I_DIM + n0 + b_n[q];
            b0P[q] = *(const float4*)(wp);
            b1P[q] = *(const float4*)(wp + I_DIM);
        }
    };
    auto store_stage = [&](int p) {
#pragma unroll
        for (int q = 0; q < 2; ++q) {
            u32* ap = (u32*)&Alds[p][a_r[q] * ASTR + a_c4[q]];
            ap[0] = pack_bf16(aP[q].x, aP[q].y);
            ap[1] = pack_bf16(aP[q].z, aP[q].w);
        }
#pragma unroll
        for (int q = 0; q < 2; ++q)
#pragma unroll
            for (int j = 0; j < 4; ++j)
                *(u32*)&Blds[p][(b_n[q] + j) * BSTR + b_k[q]] =
                    pack_bf16((&b0P[q].x)[j], (&b1P[q].x)[j]);
    };
    auto compute = [&](int p) {
        v16bf a0  = load_a_frag(&Alds[p][(wm * 32 +  0) * ASTR], lane);
        v16bf a1  = load_a_frag(&Alds[p][(wm * 32 + 16) * ASTR], lane);
        v16bf bb0 = load_b_frag(&Blds[p][(wn * 32 +  0) * BSTR], lane);
        v16bf bb1 = load_b_frag(&Blds[p][(wn * 32 + 16) * BSTR], lane);
        acc[0][0] = __builtin_amdgcn_wmma_f32_16x16x32_bf16(false, a0, false, bb0, (short)0, acc[0][0], false, false);
        acc[0][1] = __builtin_amdgcn_wmma_f32_16x16x32_bf16(false, a0, false, bb1, (short)0, acc[0][1], false, false);
        acc[1][0] = __builtin_amdgcn_wmma_f32_16x16x32_bf16(false, a1, false, bb0, (short)0, acc[1][0], false, false);
        acc[1][1] = __builtin_amdgcn_wmma_f32_16x16x32_bf16(false, a1, false, bb1, (short)0, acc[1][1], false, false);
    };

    const int NK = H_DIM / BK;   // 40
    load_regs(0);
    store_stage(0);
    load_regs(BK);
    __syncthreads();
#pragma unroll 2
    for (int kt = 0; kt < NK; ++kt) {
        int cur = kt & 1;
        if (kt + 1 < NK) store_stage(cur ^ 1);      // stage k+1 into other buffer
        if (kt + 2 < NK) load_regs((kt + 2) * BK);  // prefetch k+2
        compute(cur);
        __syncthreads();                            // single barrier per step
    }

    // epilogue: bias + exact GELU -> bf16 Hbuf
    const int lr = (lane >> 4) * 8;
    const int lc = lane & 15;
#pragma unroll
    for (int i = 0; i < 2; ++i)
#pragma unroll
    for (int j = 0; j < 2; ++j)
#pragma unroll
    for (int el = 0; el < 8; ++el) {
        int gr = m0 + wm * 32 + i * 16 + lr + el;
        int gn = n0 + wn * 32 + j * 16 + lc;
        if (gr < rows) {
            float v = gelu_exact(acc[i][j][el] + bias[gn]);
            Hbuf[(hbase + gr) * (size_t)I_DIM + gn] =
                __builtin_bit_cast(u16, (__bf16)v);
        }
    }
}

// ---------------- GEMM2: out (+)= gate * (Hbuf @ W2 + b2) [+ x] ----------------
template <bool ROUTED>
__global__ __launch_bounds__(256)
void gemm2_kernel(const u16*   __restrict__ Hbuf,  // bf16 bits [rows, I]
                  const float* __restrict__ W2,    // [(E,)I, H]
                  const float* __restrict__ B2,    // [(E,)H]
                  const float* __restrict__ X,     // residual (shared path)
                  float*       __restrict__ Out,   // [NTOK, H]
                  const int*   __restrict__ tok_list,
                  const float* __restrict__ gate_list,
                  const int*   __restrict__ cnt,
                  const int*   __restrict__ offsets)
{
    const int e    = ROUTED ? (int)blockIdx.z : 0;
    const int rows = ROUTED ? cnt[e] : NTOK;
    const int m0   = blockIdx.x * BM;
    if (m0 >= rows) return;
    const int n0   = blockIdx.y * BN;

    const float* W    = W2 + (size_t)e * I_DIM * H_DIM;
    const float* bias = B2 + (size_t)e * H_DIM;
    const size_t hbase = ROUTED ? (size_t)offsets[e] : 0;

    __shared__ __align__(16) __bf16 Alds[2][BM * ASTR];
    __shared__ __align__(16) __bf16 Blds[2][BN * BSTR];

    const int tid  = threadIdx.x;
    const int lane = tid & 31;
    const int wave = tid >> 5;
    const int wm   = wave >> 2;
    const int wn   = wave & 3;

    int  a_r[2], a_c4[2];
    const u16* a_ptr[2];
    bool a_v[2];
#pragma unroll
    for (int q = 0; q < 2; ++q) {
        int idx = tid + q * 256;
        a_r[q]  = idx >> 3;
        a_c4[q] = (idx & 7) * 4;
        int gr  = m0 + a_r[q];
        a_v[q]  = gr < rows;
        a_ptr[q] = Hbuf + (hbase + (a_v[q] ? gr : 0)) * (size_t)I_DIM + a_c4[q];
    }
    int b_k[2], b_n[2];
#pragma unroll
    for (int q = 0; q < 2; ++q) {
        int idx = tid + q * 256;
        b_n[q] = (idx & 31) * 4;
        b_k[q] = (idx >> 5) * 2;
    }

    uint2  aP[2];
    float4 b0P[2], b1P[2];
    f32x8 acc[2][2];
#pragma unroll
    for (int i = 0; i < 2; ++i)
#pragma unroll
        for (int j = 0; j < 2; ++j)
            acc[i][j] = (f32x8){0.f,0.f,0.f,0.f,0.f,0.f,0.f,0.f};

    auto load_regs = [&](int K0) {
#pragma unroll
        for (int q = 0; q < 2; ++q)
            aP[q] = a_v[q] ? *(const uint2*)(a_ptr[q] + K0) : make_uint2(0u, 0u);
#pragma unroll
        for (int q = 0; q < 2; ++q) {
            const float* wp = W + (size_t)(K0 + b_k[q]) * H_DIM + n0 + b_n[q];
            b0P[q] = *(const float4*)(wp);
            b1P[q] = *(const float4*)(wp + H_DIM);
        }
    };
    auto store_stage = [&](int p) {
#pragma unroll
        for (int q = 0; q < 2; ++q)
            *(uint2*)&Alds[p][a_r[q] * ASTR + a_c4[q]] = aP[q];   // already bf16
#pragma unroll
        for (int q = 0; q < 2; ++q)
#pragma unroll
            for (int j = 0; j < 4; ++j)
                *(u32*)&Blds[p][(b_n[q] + j) * BSTR + b_k[q]] =
                    pack_bf16((&b0P[q].x)[j], (&b1P[q].x)[j]);
    };
    auto compute = [&](int p) {
        v16bf a0  = load_a_frag(&Alds[p][(wm * 32 +  0) * ASTR], lane);
        v16bf a1  = load_a_frag(&Alds[p][(wm * 32 + 16) * ASTR], lane);
        v16bf bb0 = load_b_frag(&Blds[p][(wn * 32 +  0) * BSTR], lane);
        v16bf bb1 = load_b_frag(&Blds[p][(wn * 32 + 16) * BSTR], lane);
        acc[0][0] = __builtin_amdgcn_wmma_f32_16x16x32_bf16(false, a0, false, bb0, (short)0, acc[0][0], false, false);
        acc[0][1] = __builtin_amdgcn_wmma_f32_16x16x32_bf16(false, a0, false, bb1, (short)0, acc[0][1], false, false);
        acc[1][0] = __builtin_amdgcn_wmma_f32_16x16x32_bf16(false, a1, false, bb0, (short)0, acc[1][0], false, false);
        acc[1][1] = __builtin_amdgcn_wmma_f32_16x16x32_bf16(false, a1, false, bb1, (short)0, acc[1][1], false, false);
    };

    const int NK = I_DIM / BK;
    load_regs(0);
    store_stage(0);
    load_regs(BK);
    __syncthreads();
#pragma unroll 2
    for (int kt = 0; kt < NK; ++kt) {
        int cur = kt & 1;
        if (kt + 1 < NK) store_stage(cur ^ 1);
        if (kt + 2 < NK) load_regs((kt + 2) * BK);
        compute(cur);
        __syncthreads();
    }

    const int lr = (lane >> 4) * 8;
    const int lc = lane & 15;
#pragma unroll
    for (int i = 0; i < 2; ++i)
#pragma unroll
    for (int j = 0; j < 2; ++j)
#pragma unroll
    for (int el = 0; el < 8; ++el) {
        int gr = m0 + wm * 32 + i * 16 + lr + el;
        int gn = n0 + wn * 32 + j * 16 + lc;
        if (gr < rows) {
            if (ROUTED) {
                int   tok = tok_list[e * NTOK + gr];
                float g   = gate_list[e * NTOK + gr];
                atomicAdd(&Out[(size_t)tok * H_DIM + gn],
                          g * (acc[i][j][el] + bias[gn]));
            } else {
                size_t oi = (size_t)gr * H_DIM + gn;
                Out[oi] = acc[i][j][el] + bias[gn] + X[oi];  // shared + residual
            }
        }
    }
}

// ---------------- launcher ----------------
extern "C" void kernel_launch(void* const* d_in, const int* in_sizes, int n_in,
                              void* d_out, int out_size, void* d_ws, size_t ws_size,
                              hipStream_t stream) {
    const float* x   = (const float*)d_in[0];
    const float* w1s = (const float*)d_in[1];
    const float* b1s = (const float*)d_in[2];
    const float* w2s = (const float*)d_in[3];
    const float* b2s = (const float*)d_in[4];
    const float* rw  = (const float*)d_in[5];
    const float* rb  = (const float*)d_in[6];
    const float* w1  = (const float*)d_in[7];
    const float* b1  = (const float*)d_in[8];
    const float* w2  = (const float*)d_in[9];
    const float* b2  = (const float*)d_in[10];
    float* out = (float*)d_out;

    // workspace carve-up (~86 MB)
    char* w = (char*)d_ws;
    int*   cnt       = (int*)w;            w += 256;
    int*   offsets   = (int*)w;            w += 256;
    int*   tok_list  = (int*)w;            w += sizeof(int)   * (size_t)E_NUM * NTOK;
    float* gate_list = (float*)w;          w += sizeof(float) * (size_t)E_NUM * NTOK;
    u16*   Hshared   = (u16*)w;            w += sizeof(u16)   * (size_t)NTOK * I_DIM;
    u16*   Hrouted   = (u16*)w;            // exactly 7*NTOK rows (top-7 per token)

    hipMemsetAsync(cnt, 0, 256, stream);

    router_kernel<<<NTOK, 64, 0, stream>>>(x, rw, rb, cnt, tok_list, gate_list);
    offsets_kernel<<<1, 32, 0, stream>>>(cnt, offsets);

    // shared expert: out = x + ffn_shared(x)
    gemm1_kernel<false><<<dim3(NTOK / BM, I_DIM / BN), 256, 0, stream>>>(
        x, w1s, b1s, Hshared, nullptr, nullptr, nullptr);
    gemm2_kernel<false><<<dim3(NTOK / BM, H_DIM / BN), 256, 0, stream>>>(
        Hshared, w2s, b2s, x, out, nullptr, nullptr, nullptr, nullptr);

    // routed experts: out += gate_e * ffn_e(x) over gathered token lists
    gemm1_kernel<true><<<dim3(NTOK / BM, I_DIM / BN, E_NUM), 256, 0, stream>>>(
        x, w1, b1, Hrouted, tok_list, cnt, offsets);
    gemm2_kernel<true><<<dim3(NTOK / BM, H_DIM / BN, E_NUM), 256, 0, stream>>>(
        Hrouted, w2, b2, x, out, tok_list, gate_list, cnt, offsets);
}
